// CausalWanModel_12438225290014
// MI455X (gfx1250) — compile-verified, matrix operand order checked by reading, not run on hardware
//
#include <hip/hip_runtime.h>
#include <hip/hip_bf16.h>
#include <stdint.h>

#define D_MODEL 1536
#define T_SEQ   3120
#define NHEADS  12
#define HDIM    128
#define NQB     (T_SEQ / 16)     /* 195 query blocks of 16 rows */

typedef __bf16 bf16_t;
typedef bf16_t v16bf __attribute__((ext_vector_type(16)));
typedef bf16_t v8bf  __attribute__((ext_vector_type(8)));
typedef float  v8f   __attribute__((ext_vector_type(8)));
typedef unsigned int uint32x4 __attribute__((ext_vector_type(4)));
typedef int int32x4 __attribute__((ext_vector_type(4)));
typedef int int32x8 __attribute__((ext_vector_type(8)));

// ---------------------------------------------------------------------------
// helpers
// ---------------------------------------------------------------------------
static __device__ __forceinline__ v8f vzero8() {
  v8f r;
#pragma unroll
  for (int i = 0; i < 8; ++i) r[i] = 0.0f;
  return r;
}

// Load one 16x32 (bf16) WMMA A/B operand for this lane.
// Per ISA 7.12.2: lane (idx = lane&15, khalf = lane>>4) holds two contiguous
// 8-element chunks of its row/column at k = k0 + 8*khalf and k0 + 8*khalf + 16.
static __device__ __forceinline__ v16bf ld_tile(const bf16_t* __restrict__ row,
                                                int k0, int kh) {
  const v8bf c1 = *(const v8bf*)(row + k0 + 8 * kh);
  const v8bf c2 = *(const v8bf*)(row + k0 + 8 * kh + 16);
  v16bf r;
#pragma unroll
  for (int i = 0; i < 8; ++i) { r[i] = c1[i]; r[i + 8] = c2[i]; }
  return r;
}

static __device__ __forceinline__ v8f wmma_bf16(v16bf a, v16bf b, v8f c) {
  return __builtin_amdgcn_wmma_f32_16x16x32_bf16(false, a, false, b,
                                                 (short)0, c, false, false);
}

// Issue a TDM 2D tile load (bf16 elements) global -> LDS, per ISA ch.8 D#.
// rows x cols tile, row-major global with row_stride (elements), packed LDS.
static __device__ __forceinline__ void tdm_load_2d(const void* gptr,
                                                   unsigned lds_off, int rows,
                                                   int cols,
                                                   int row_stride_elems) {
  const unsigned long long ga = (unsigned long long)(size_t)gptr;
  uint32x4 g0;
  g0[0] = 1u;                                     // count=1, user descriptor
  g0[1] = lds_off;                                // lds_addr (bytes)
  g0[2] = (unsigned)(ga & 0xffffffffu);           // global_addr[31:0]
  g0[3] = (unsigned)((ga >> 32) & 0x01ffffffu) |  // global_addr[56:32]
          (2u << 30);                             // type=2 ("image")
  int32x8 g1;
  g1[0] = (1 << 16);                              // wg_mask=0, data_size=1 (2B)
  g1[1] = (cols & 0xffff) << 16;                  // tensor_dim0[15:0]
  g1[2] = ((cols >> 16) & 0xffff) |               // tensor_dim0[31:16]
          ((rows & 0xffff) << 16);                // tensor_dim1[15:0]
  g1[3] = ((rows >> 16) & 0xffff) |               // tensor_dim1[31:16]
          ((cols & 0xffff) << 16);                // tile_dim0
  g1[4] = (rows & 0xffff);                        // tile_dim1; tile_dim2=0
  g1[5] = row_stride_elems;                       // tensor_dim0_stride[31:0]
  g1[6] = 0;                                      // stride hi / dim1_stride lo
  g1[7] = 0;
  int32x4 gz;
  gz[0] = gz[1] = gz[2] = gz[3] = 0;
#if __clang_major__ >= 23
  int32x8 gz8;
#pragma unroll
  for (int i = 0; i < 8; ++i) gz8[i] = 0;
  __builtin_amdgcn_tensor_load_to_lds(g0, g1, gz, gz, gz8, 0);
#else
  __builtin_amdgcn_tensor_load_to_lds(g0, g1, gz, gz, 0);
#endif
}

// ---------------------------------------------------------------------------
// prep: fp32 -> bf16 conversion / weight transpose (N-major bf16)
// ---------------------------------------------------------------------------
__global__ __launch_bounds__(256) void cvt_x_kernel(const float* __restrict__ x,
                                                    bf16_t* __restrict__ xb, int n) {
  int i = blockIdx.x * 256 + threadIdx.x;
  if (i < n) xb[i] = (bf16_t)x[i];
}

__global__ __launch_bounds__(256) void transpose_kernel(const float* __restrict__ W,
                                                        bf16_t* __restrict__ wT) {
  __shared__ float tile[32][33];
  const int k0 = blockIdx.x * 32, n0 = blockIdx.y * 32;
  const int tx = threadIdx.x & 31, ty = threadIdx.x >> 5;
#pragma unroll
  for (int r = ty; r < 32; r += 8)
    tile[r][tx] = W[(size_t)(k0 + r) * D_MODEL + n0 + tx];
  __syncthreads();
#pragma unroll
  for (int r = ty; r < 32; r += 8)
    wT[(size_t)(n0 + r) * D_MODEL + k0 + tx] = (bf16_t)tile[tx][r];
}

// ---------------------------------------------------------------------------
// fused QKV projection + RMSNorm + RoPE (A operand staged in LDS via TDM)
// ---------------------------------------------------------------------------
static __device__ __forceinline__ void proj_qk(
    const bf16_t* s_xa, const bf16_t* __restrict__ wT,
    const float* __restrict__ bias, const float* __restrict__ g,
    const float* __restrict__ fc, const float* __restrict__ fs,
    bf16_t* __restrict__ outp, int t0, float* s_sum, bf16_t* s_buf) {
  const int tid = threadIdx.x;
  const int wave = tid >> 5, lane = tid & 31;
  const int m = lane & 15, kh = lane >> 4;
  if (tid < 16) s_sum[tid] = 0.0f;
  __syncthreads();
  const bf16_t* arow = s_xa + (size_t)m * D_MODEL;
  for (int tp = 0; tp < 6; ++tp) {
    const int n0 = (wave * 12 + tp * 2) * 16;
    const int col0 = n0 + m, col1 = n0 + 16 + m;
    const bf16_t* brow0 = wT + (size_t)col0 * D_MODEL;
    const bf16_t* brow1 = wT + (size_t)col1 * D_MODEL;
    if (tp + 1 < 6)
      __builtin_prefetch(wT + (size_t)(col0 + 32) * D_MODEL, 0, 1);
    v8f acc0 = vzero8(), acc1 = vzero8();
    for (int k0 = 0; k0 < D_MODEL; k0 += 32) {
      const v16bf a = ld_tile(arow, k0, kh);
      acc0 = wmma_bf16(a, ld_tile(brow0, k0, kh), acc0);
      acc1 = wmma_bf16(a, ld_tile(brow1, k0, kh), acc1);
    }
#pragma unroll
    for (int u = 0; u < 2; ++u) {
      const int col = u ? col1 : col0;
      const v8f acc = u ? acc1 : acc0;
      const float bb = bias[col];
#pragma unroll
      for (int j = 0; j < 8; ++j) {
        const float v = acc[j] + bb;               // C row = j + 8*kh
        atomicAdd(&s_sum[j + 8 * kh], v * v);      // ds_add_f32
        s_buf[(size_t)(j + 8 * kh) * D_MODEL + col] = (bf16_t)v;
      }
    }
  }
  __syncthreads();
  // RMSNorm + RoPE + scatter to head-major [H][T][hd]
  for (int p = tid; p < 16 * (D_MODEL / 2); p += 256) {
    const int row = p / (D_MODEL / 2);
    const int d = (p % (D_MODEL / 2)) * 2;
    const int head = d >> 7, dh = d & 127, fj = dh >> 1;
    const int t = t0 + row;
    const float rms = rsqrtf(s_sum[row] * (1.0f / D_MODEL) + 1e-6f);
    const float a = (float)s_buf[(size_t)row * D_MODEL + d]     * rms * g[d];
    const float b = (float)s_buf[(size_t)row * D_MODEL + d + 1] * rms * g[d + 1];
    const float c = fc[t * (HDIM / 2) + fj];
    const float s = fs[t * (HDIM / 2) + fj];
    bf16_t* dst = outp + ((size_t)head * T_SEQ + t) * HDIM + dh;
    dst[0] = (bf16_t)(a * c - b * s);
    dst[1] = (bf16_t)(a * s + b * c);
  }
  __syncthreads();
}

__global__ __launch_bounds__(256) void qkv_kernel(
    const bf16_t* __restrict__ xb,
    const bf16_t* __restrict__ wqT, const bf16_t* __restrict__ wkT,
    const bf16_t* __restrict__ wvT,
    const float* __restrict__ bq, const float* __restrict__ bk,
    const float* __restrict__ bvv,
    const float* __restrict__ gq, const float* __restrict__ gk,
    const float* __restrict__ fc, const float* __restrict__ fs,
    bf16_t* __restrict__ qh, bf16_t* __restrict__ khd,
    bf16_t* __restrict__ vT) {
  __shared__ __align__(16) bf16_t s_xa[16 * D_MODEL];   // 48 KB, TDM-staged A
  __shared__ float s_sum[16];
  __shared__ __align__(16) bf16_t s_buf[16 * D_MODEL];  // 48 KB
  const int t0 = blockIdx.x * 16;
  const int tid = threadIdx.x;
  const int wave = tid >> 5, lane = tid & 31;
  const int m = lane & 15, kh = lane >> 4;

  if (wave == 0) {
    tdm_load_2d(xb + (size_t)t0 * D_MODEL, (unsigned)(size_t)(void*)s_xa, 16,
                D_MODEL, D_MODEL);
    __builtin_amdgcn_s_wait_tensorcnt(0);
  }
  __syncthreads();

  proj_qk(s_xa, wqT, bq, gq, fc, fs, qh, t0, s_sum, s_buf);
  proj_qk(s_xa, wkT, bk, gk, fc, fs, khd, t0, s_sum, s_buf);

  // V: no norm/rope; store transposed [H][hd][T] so P@V B-columns are contiguous
  const bf16_t* arow = s_xa + (size_t)m * D_MODEL;
  for (int tp = 0; tp < 6; ++tp) {
    const int n0 = (wave * 12 + tp * 2) * 16;
    const int col0 = n0 + m, col1 = n0 + 16 + m;
    const bf16_t* brow0 = wvT + (size_t)col0 * D_MODEL;
    const bf16_t* brow1 = wvT + (size_t)col1 * D_MODEL;
    v8f acc0 = vzero8(), acc1 = vzero8();
    for (int k0 = 0; k0 < D_MODEL; k0 += 32) {
      const v16bf a = ld_tile(arow, k0, kh);
      acc0 = wmma_bf16(a, ld_tile(brow0, k0, kh), acc0);
      acc1 = wmma_bf16(a, ld_tile(brow1, k0, kh), acc1);
    }
#pragma unroll
    for (int u = 0; u < 2; ++u) {
      const int col = u ? col1 : col0;
      const v8f acc = u ? acc1 : acc0;
      const float bb = bvv[col];
      v8bf pk;
#pragma unroll
      for (int j = 0; j < 8; ++j) pk[j] = (bf16_t)(acc[j] + bb);
      const int head = col >> 7, dh = col & 127;
      // rows t0+8*kh .. +7 contiguous along T in vT -> one 16B store
      *(v8bf*)(vT + ((size_t)(head * HDIM + dh)) * T_SEQ + t0 + 8 * kh) = pk;
    }
  }
}

// ---------------------------------------------------------------------------
// flash-style block-causal attention; one wave per (head, 16-query block)
// ---------------------------------------------------------------------------
__global__ __launch_bounds__(32) void attn_kernel(
    const bf16_t* __restrict__ qh, const bf16_t* __restrict__ khd,
    const bf16_t* __restrict__ vT, const int* __restrict__ fsl,
    bf16_t* __restrict__ ao) {
  const int FRAME = fsl[0];
  const int head = blockIdx.x / NQB;
  const int t0 = (blockIdx.x % NQB) * 16;
  const int lane = threadIdx.x;
  const int m = lane & 15, kh = lane >> 4;
  __shared__ __align__(16) bf16_t pbuf[16 * 32];

  const bf16_t* qrow = qh + ((size_t)head * T_SEQ + (t0 + m)) * HDIM;
  v16bf qa[4];
#pragma unroll
  for (int kk = 0; kk < 4; ++kk) qa[kk] = ld_tile(qrow, kk * 32, kh);

  int rlim[8];
#pragma unroll
  for (int j = 0; j < 8; ++j)
    rlim[j] = (((t0 + j + 8 * kh) / FRAME) + 1) * FRAME;
  const int sEnd = (((t0 + 15) / FRAME) + 1) * FRAME;

  float mrun[8], lrun[8];
  v8f oacc[8];
#pragma unroll
  for (int j = 0; j < 8; ++j) { mrun[j] = -1e30f; lrun[j] = 0.0f; }
#pragma unroll
  for (int n = 0; n < 8; ++n) oacc[n] = vzero8();

  const float scale = 0.08838834764831845f;   // 1/sqrt(128)

  for (int s0 = 0; s0 < sEnd; s0 += 32) {
    const int sA = s0 + m;
    const int sB = s0 + 16 + m;
    const bf16_t* kr0 =
        khd + ((size_t)head * T_SEQ + (sA < T_SEQ ? sA : T_SEQ - 1)) * HDIM;
    const bf16_t* kr1 =
        khd + ((size_t)head * T_SEQ + (sB < T_SEQ ? sB : T_SEQ - 1)) * HDIM;
    v8f sc0 = vzero8(), sc1 = vzero8();
#pragma unroll
    for (int kk = 0; kk < 4; ++kk) {
      sc0 = wmma_bf16(qa[kk], ld_tile(kr0, kk * 32, kh), sc0);
      sc1 = wmma_bf16(qa[kk], ld_tile(kr1, kk * 32, kh), sc1);
    }
    float p0[8], p1[8], corr[8];
#pragma unroll
    for (int j = 0; j < 8; ++j) {
      const float v0 = (sA < rlim[j]) ? sc0[j] * scale : -3.0e38f;
      const float v1 = (sB < rlim[j]) ? sc1[j] * scale : -3.0e38f;
      float mj = fmaxf(v0, v1);
#pragma unroll
      for (int off = 1; off < 16; off <<= 1)
        mj = fmaxf(mj, __shfl_xor(mj, off, 32));   // row lives on 16 lanes
      const float mnew = fmaxf(mrun[j], mj);
      corr[j] = __expf(mrun[j] - mnew);
      mrun[j] = mnew;
      const float e0 = __expf(v0 - mnew);
      const float e1 = __expf(v1 - mnew);
      float rs = e0 + e1;
#pragma unroll
      for (int off = 1; off < 16; off <<= 1) rs += __shfl_xor(rs, off, 32);
      lrun[j] = lrun[j] * corr[j] + rs;
      p0[j] = e0;
      p1[j] = e1;
    }
#pragma unroll
    for (int n = 0; n < 8; ++n)
#pragma unroll
      for (int j = 0; j < 8; ++j) oacc[n][j] *= corr[j];

    // repack P (C layout) -> A layout through LDS
#pragma unroll
    for (int j = 0; j < 8; ++j) {
      pbuf[(j + 8 * kh) * 32 + m] = (bf16_t)p0[j];
      pbuf[(j + 8 * kh) * 32 + 16 + m] = (bf16_t)p1[j];
    }
    __syncthreads();
    const v16bf pa = ld_tile(pbuf + m * 32, 0, kh);
#pragma unroll
    for (int n = 0; n < 8; ++n) {
      const int d = n * 16 + m;                 // B column = output dim
      const bf16_t* vrow = vT + ((size_t)head * HDIM + d) * T_SEQ;
      int c1 = s0 + 8 * kh;       if (c1 > T_SEQ - 8) c1 = T_SEQ - 8;
      int c2 = s0 + 8 * kh + 16;  if (c2 > T_SEQ - 8) c2 = T_SEQ - 8;
      const v8bf b1 = *(const v8bf*)(vrow + c1);
      const v8bf b2 = *(const v8bf*)(vrow + c2);
      v16bf vb;
#pragma unroll
      for (int i = 0; i < 8; ++i) { vb[i] = b1[i]; vb[i + 8] = b2[i]; }
      oacc[n] = wmma_bf16(pa, vb, oacc[n]);
    }
    __syncthreads();
  }

#pragma unroll
  for (int n = 0; n < 8; ++n) {
    const int col = head * HDIM + n * 16 + m;
#pragma unroll
    for (int j = 0; j < 8; ++j) {
      const float o = oacc[n][j] / lrun[j];
      ao[(size_t)(t0 + j + 8 * kh) * D_MODEL + col] = (bf16_t)o;
    }
  }
}

// ---------------------------------------------------------------------------
// output projection: out = ao @ Wo + bo  (fp32 out), A staged via TDM
// ---------------------------------------------------------------------------
__global__ __launch_bounds__(256) void oproj_kernel(
    const bf16_t* __restrict__ ao, const bf16_t* __restrict__ woT,
    const float* __restrict__ bo, float* __restrict__ out) {
  __shared__ __align__(16) bf16_t s_a[16 * D_MODEL];    // 48 KB
  const int t0 = blockIdx.x * 16;
  const int tid = threadIdx.x, wave = tid >> 5, lane = tid & 31;
  const int m = lane & 15, kh = lane >> 4;

  if (wave == 0) {
    tdm_load_2d(ao + (size_t)t0 * D_MODEL, (unsigned)(size_t)(void*)s_a, 16,
                D_MODEL, D_MODEL);
    __builtin_amdgcn_s_wait_tensorcnt(0);
  }
  __syncthreads();

  const bf16_t* arow = s_a + (size_t)m * D_MODEL;
  for (int tp = 0; tp < 6; ++tp) {
    const int n0 = (wave * 12 + tp * 2) * 16;
    const int col0 = n0 + m, col1 = n0 + 16 + m;
    const bf16_t* brow0 = woT + (size_t)col0 * D_MODEL;
    const bf16_t* brow1 = woT + (size_t)col1 * D_MODEL;
    if (tp + 1 < 6)
      __builtin_prefetch(woT + (size_t)(col0 + 32) * D_MODEL, 0, 1);
    v8f acc0 = vzero8(), acc1 = vzero8();
    for (int k0 = 0; k0 < D_MODEL; k0 += 32) {
      const v16bf a = ld_tile(arow, k0, kh);
      acc0 = wmma_bf16(a, ld_tile(brow0, k0, kh), acc0);
      acc1 = wmma_bf16(a, ld_tile(brow1, k0, kh), acc1);
    }
#pragma unroll
    for (int u = 0; u < 2; ++u) {
      const int col = u ? col1 : col0;
      const v8f acc = u ? acc1 : acc0;
      const float bb = bo[col];
#pragma unroll
      for (int j = 0; j < 8; ++j)
        out[(size_t)(t0 + j + 8 * kh) * D_MODEL + col] = acc[j] + bb;
    }
  }
}

// ---------------------------------------------------------------------------
extern "C" void kernel_launch(void* const* d_in, const int* in_sizes, int n_in,
                              void* d_out, int out_size, void* d_ws,
                              size_t ws_size, hipStream_t stream) {
  const float* x  = (const float*)d_in[0];
  const float* fc = (const float*)d_in[1];
  const float* fs = (const float*)d_in[2];
  const float* Wq = (const float*)d_in[3];  const float* bq = (const float*)d_in[4];
  const float* Wk = (const float*)d_in[5];  const float* bk = (const float*)d_in[6];
  const float* Wv = (const float*)d_in[7];  const float* bv = (const float*)d_in[8];
  const float* Wo = (const float*)d_in[9];  const float* bo = (const float*)d_in[10];
  const float* gq = (const float*)d_in[11]; const float* gk = (const float*)d_in[12];
  const int* fsl  = (const int*)d_in[13];
  float* out = (float*)d_out;

  char* ws = (char*)d_ws;
  size_t off = 0;
  auto alloc = [&](size_t bytes) -> char* {
    char* p = ws + off;
    off += (bytes + 255) & ~(size_t)255;
    return p;
  };
  const size_t td = (size_t)T_SEQ * D_MODEL * sizeof(bf16_t);
  const size_t dd = (size_t)D_MODEL * D_MODEL * sizeof(bf16_t);
  bf16_t* xb  = (bf16_t*)alloc(td);
  bf16_t* wqT = (bf16_t*)alloc(dd);
  bf16_t* wkT = (bf16_t*)alloc(dd);
  bf16_t* wvT = (bf16_t*)alloc(dd);
  bf16_t* woT = (bf16_t*)alloc(dd);
  bf16_t* qh  = (bf16_t*)alloc(td);   // [H][T][hd]
  bf16_t* khd = (bf16_t*)alloc(td);   // [H][T][hd]
  bf16_t* vT  = (bf16_t*)alloc(td);   // [H][hd][T]
  bf16_t* ao  = (bf16_t*)alloc(td);   // [T][D]
  if (off > ws_size) return;          // insufficient scratch; nothing we can do

  cvt_x_kernel<<<(T_SEQ * D_MODEL + 255) / 256, 256, 0, stream>>>(
      x, xb, T_SEQ * D_MODEL);
  dim3 tg(D_MODEL / 32, D_MODEL / 32);
  transpose_kernel<<<tg, 256, 0, stream>>>(Wq, wqT);
  transpose_kernel<<<tg, 256, 0, stream>>>(Wk, wkT);
  transpose_kernel<<<tg, 256, 0, stream>>>(Wv, wvT);
  transpose_kernel<<<tg, 256, 0, stream>>>(Wo, woT);

  qkv_kernel<<<NQB, 256, 0, stream>>>(xb, wqT, wkT, wvT, bq, bk, bv, gq, gk,
                                      fc, fs, qh, khd, vT);
  attn_kernel<<<NHEADS * NQB, 32, 0, stream>>>(qh, khd, vT, fsl, ao);
  oproj_kernel<<<NQB, 256, 0, stream>>>(ao, woT, bo, out);
}
